// FusionEncoder_73924977098773
// MI455X (gfx1250) — matrix operand order repaired and checked by reading, not compiled, not run on hardware
//
#include <hip/hip_runtime.h>
#include <hip/hip_bf16.h>
#include <math.h>

#define N_DOCS 131072
#define E_DIM 384
#define H_DIM 768
#define TWO_E 768

typedef __attribute__((ext_vector_type(16))) __bf16 v16bf;
typedef __attribute__((ext_vector_type(8)))  float  v8f;

// workspace layout (float indices)
#define WS_QBP    0        // 384: q_part + bp
#define WS_V      384      // 384: Wp[:,E:]^T @ Wa (f32)
#define WS_C      768      // scalar score bias
#define WS_M      769      // global max
#define WS_INVS   770      // 1/sum(exp)
#define WS_PMAX   772      // 1024 block-max partials
#define WS_PSUM   1796     // 512 block-sum partials
#define WS_DBAR   2308     // 384: sum_n w_n * doc[n]
#define WS_BPLANE 2696     // 12*32*16 bf16 = 3072 float slots; 2696*4=10784, 32B aligned

__device__ __forceinline__ __bf16 tobf(float x) { return static_cast<__bf16>(x); }

// ---------------- K1: qbp, v, c, zero dbar, build WMMA B-fragment plane ----------------
__global__ __launch_bounds__(384)
void k_prep(const float* __restrict__ q, const float* __restrict__ Wp,
            const float* __restrict__ bp, const float* __restrict__ Wa,
            const float* __restrict__ ba, float* __restrict__ ws) {
  __shared__ float red[384];
  __shared__ float sv[384];
  const int t = threadIdx.x;
  // qbp[t] = Wp[t, :E] . q + bp[t]
  float acc = bp[t];
  for (int k = 0; k < E_DIM; ++k) acc += Wp[t * TWO_E + k] * q[k];
  ws[WS_QBP + t] = acc;
  // v[t] = sum_j Wa[j] * Wp[j, E + t]   (coalesced across t)
  float vv = 0.f;
  for (int j = 0; j < E_DIM; ++j) vv += Wa[j] * Wp[j * TWO_E + E_DIM + t];
  ws[WS_V + t] = vv;
  sv[t] = vv;
  ws[WS_DBAR + t] = 0.f;
  red[t] = acc * Wa[t];
  __syncthreads();
  // B-fragment plane: for step s (k0=32s), lane L, elem i:
  //   lane L<16 covers K=0..7,16..23 (col N=L), lane>=16 covers K=8..15,24..31 (col N=L-16);
  //   only column N==0 carries v, everything else is zero.
  __bf16* __restrict__ bplane = (__bf16*)(ws + WS_BPLANE);
  for (int idx = t; idx < 12 * 32 * 16; idx += 384) {
    const int i = idx & 15;
    const int L = (idx >> 4) & 31;
    const int s = idx >> 9;
    const float val = ((L & 15) == 0) ? sv[s * 32 + (i >> 3) * 16 + (L >> 4) * 8 + (i & 7)] : 0.f;
    bplane[idx] = tobf(val);
  }
  // c = qbp . Wa + ba
  if (t < 128) red[t] = red[t] + red[t + 128] + red[t + 256];
  __syncthreads();
  for (int s = 64; s > 0; s >>= 1) {
    if (t < s) red[t] += red[t + s];
    __syncthreads();
  }
  if (t == 0) ws[WS_C] = red[0] + ba[0];
}

// ---------------- K2: scores via BF16 WMMA (HBM-bound streaming pass) ----------------
__global__ __launch_bounds__(256)
void k_scores(const float* __restrict__ doc, const float* __restrict__ ws,
              float* __restrict__ sout, float* __restrict__ pmax) {
  __shared__ float smax[256];
  const int tid  = threadIdx.x;
  const int wave = tid >> 5;
  const int lane = tid & 31;
  const int row  = lane & 15;       // M row within tile
  const int kh   = lane >> 4;       // K half selector
  const int doc0 = blockIdx.x * 128 + wave * 16;
  const float*  __restrict__ drow   = doc + (size_t)(doc0 + row) * E_DIM;
  const __bf16* __restrict__ bplane = (const __bf16*)(ws + WS_BPLANE);
  const float cbias = ws[WS_C];

  v8f acc0 = {0.f,0.f,0.f,0.f,0.f,0.f,0.f,0.f};
  v8f acc1 = {0.f,0.f,0.f,0.f,0.f,0.f,0.f,0.f};
#pragma unroll
  for (int s = 0; s < 12; ++s) {
    const int k0 = s * 32;
    // A fragment: 16x32 bf16. lane<16: K=0..7,16..23 ; lane>=16: K=8..15,24..31
    const int kLo = k0 + kh * 8;
    const int kHi = kLo + 16;
    float4 x0 = *(const float4*)(drow + kLo);
    float4 x1 = *(const float4*)(drow + kLo + 4);
    float4 y0 = *(const float4*)(drow + kHi);
    float4 y1 = *(const float4*)(drow + kHi + 4);
    v16bf a;
    a[0]=tobf(x0.x);  a[1]=tobf(x0.y);  a[2]=tobf(x0.z);  a[3]=tobf(x0.w);
    a[4]=tobf(x1.x);  a[5]=tobf(x1.y);  a[6]=tobf(x1.z);  a[7]=tobf(x1.w);
    a[8]=tobf(y0.x);  a[9]=tobf(y0.y);  a[10]=tobf(y0.z); a[11]=tobf(y0.w);
    a[12]=tobf(y1.x); a[13]=tobf(y1.y); a[14]=tobf(y1.z); a[15]=tobf(y1.w);
    // Precomputed B fragment: one aligned 32B lane-coalesced load, no masking.
    v16bf b = *(const v16bf*)(bplane + (size_t)s * 512 + lane * 16);
    if (s & 1)
      acc1 = __builtin_amdgcn_wmma_f32_16x16x32_bf16(
          false, a, false, b, (short)0, acc1, false, false);
    else
      acc0 = __builtin_amdgcn_wmma_f32_16x16x32_bf16(
          false, a, false, b, (short)0, acc0, false, false);
  }

  // C column 0: lane 0 holds M=0..7, lane 16 holds M=8..15
  float lmax = -INFINITY;
  if (row == 0) {
    const int base = doc0 + kh * 8;
#pragma unroll
    for (int r = 0; r < 8; ++r) {
      float sc = acc0[r] + acc1[r] + cbias;
      sout[base + r] = sc;
      lmax = fmaxf(lmax, sc);
    }
  }
  smax[tid] = lmax;
  __syncthreads();
  for (int s = 128; s > 0; s >>= 1) {
    if (tid < s) smax[tid] = fmaxf(smax[tid], smax[tid + s]);
    __syncthreads();
  }
  if (tid == 0) pmax[blockIdx.x] = smax[0];
}

// ---------------- K3: reduce 1024 block maxes ----------------
__global__ __launch_bounds__(1024)
void k_redmax(float* __restrict__ ws) {
  __shared__ float sm[1024];
  const int t = threadIdx.x;
  sm[t] = ws[WS_PMAX + t];
  __syncthreads();
  for (int s = 512; s > 0; s >>= 1) {
    if (t < s) sm[t] = fmaxf(sm[t], sm[t + s]);
    __syncthreads();
  }
  if (t == 0) ws[WS_M] = sm[0];
}

// ---------------- K4: exp in place + block partial sums ----------------
__global__ __launch_bounds__(256)
void k_exp(float* __restrict__ sout, float* __restrict__ ws) {
  __shared__ float ss[256];
  const int t = threadIdx.x;
  const int i = blockIdx.x * 256 + t;
  const float M = ws[WS_M];
  float e = __expf(sout[i] - M);
  sout[i] = e;
  ss[t] = e;
  __syncthreads();
  for (int s = 128; s > 0; s >>= 1) {
    if (t < s) ss[t] += ss[t + s];
    __syncthreads();
  }
  if (t == 0) ws[WS_PSUM + blockIdx.x] = ss[0];
}

// ---------------- K5: reduce 512 partial sums -> 1/S ----------------
__global__ __launch_bounds__(512)
void k_redsum(float* __restrict__ ws) {
  __shared__ float ss[512];
  const int t = threadIdx.x;
  ss[t] = ws[WS_PSUM + t];
  __syncthreads();
  for (int s = 256; s > 0; s >>= 1) {
    if (t < s) ss[t] += ss[t + s];
    __syncthreads();
  }
  if (t == 0) ws[WS_INVS] = 1.0f / ss[0];
}

// ---------------- K6: weights out + dbar = sum_n w_n * doc[n] (2nd streaming pass) ----
// Blocks iterate docs in REVERSE order: after k_scores streamed doc front-to-back,
// L2 (192MB) holds the tail of the 201MB matrix; reading back-to-front turns the
// cyclic-LRU worst case into mostly L2 hits.
__global__ __launch_bounds__(384)
void k_weights_dbar(const float* __restrict__ doc, float* __restrict__ escore,
                    float* __restrict__ ws) {
  __shared__ float wsh[512];
  const int t = threadIdx.x;                                 // column owner: coalesced
  const int base = (int)(gridDim.x - 1 - blockIdx.x) * 512;  // 512 docs per block
  const float invS = ws[WS_INVS];
  for (int idx = t; idx < 512; idx += 384) {
    const float w = escore[base + idx] * invS;
    wsh[idx] = w;
    escore[base + idx] = w;                                  // final weights out
  }
  __syncthreads();
  float acc = 0.f;
  const float* __restrict__ p = doc + (size_t)base * E_DIM + t;
  for (int n = 0; n < 512; ++n) {
    __builtin_prefetch(p + (size_t)8 * E_DIM, 0, 0);         // global_prefetch_b8
    acc += wsh[n] * p[0];
    p += E_DIM;
  }
  atomicAdd(&ws[WS_DBAR + t], acc);
}

// ---------------- K7: fused = dbar @ Wpd^T + qbp ; out = Wf @ fused + bf ----------------
__global__ __launch_bounds__(768)
void k_final(const float* __restrict__ Wp, const float* __restrict__ Wf,
             const float* __restrict__ bf_, float* __restrict__ out,
             const float* __restrict__ ws) {
  __shared__ float fused[E_DIM];
  const int t = threadIdx.x;
  if (t < E_DIM) {
    float a = ws[WS_QBP + t];
    for (int k = 0; k < E_DIM; ++k)
      a += ws[WS_DBAR + k] * Wp[t * TWO_E + E_DIM + k];
    fused[t] = a;
  }
  __syncthreads();
  float o = bf_[t];
  for (int j = 0; j < E_DIM; ++j) o += Wf[t * E_DIM + j] * fused[j];
  out[t] = o;
}

extern "C" void kernel_launch(void* const* d_in, const int* in_sizes, int n_in,
                              void* d_out, int out_size, void* d_ws, size_t ws_size,
                              hipStream_t stream) {
  const float* q   = (const float*)d_in[0];
  const float* doc = (const float*)d_in[1];
  const float* Wp  = (const float*)d_in[2];
  const float* bp  = (const float*)d_in[3];
  const float* Wa  = (const float*)d_in[4];
  const float* ba  = (const float*)d_in[5];
  const float* Wf  = (const float*)d_in[6];
  const float* bf_ = (const float*)d_in[7];
  float* out = (float*)d_out;
  float* wts = out + H_DIM;      // N_DOCS region: scores -> exp -> weights
  float* ws  = (float*)d_ws;

  k_prep        <<<1, 384, 0, stream>>>(q, Wp, bp, Wa, ba, ws);
  k_scores      <<<N_DOCS / 128, 256, 0, stream>>>(doc, ws, wts, ws + WS_PMAX);
  k_redmax      <<<1, 1024, 0, stream>>>(ws);
  k_exp         <<<N_DOCS / 256, 256, 0, stream>>>(wts, ws);
  k_redsum      <<<1, 512, 0, stream>>>(ws);
  k_weights_dbar<<<N_DOCS / 512, 384, 0, stream>>>(doc, wts, ws);
  k_final       <<<1, 768, 0, stream>>>(Wp, Wf, bf_, out, ws);
}
